// LSTM_71871982731369
// MI455X (gfx1250) — compile-verified, matrix operand order checked by reading, not run on hardware
//
#include <hip/hip_runtime.h>
#include <hip/hip_bf16.h>

typedef __bf16 bf16_t;
typedef __attribute__((ext_vector_type(16))) __bf16 v16bf;
typedef __attribute__((ext_vector_type(8)))  float  v8f;

#define LSTM_T 256
#define LSTM_B 512
#define LSTM_H 512
#define LSTM_E 10
#define LSTM_C 10

// ---------------------------------------------------------------------------
// Prep: swizzle recurrent weights Wh[g][K=512][N=512] (f32) into bf16 WMMA
// B-fragments, ordered so each wave's 8 tiles are contiguous with kb next:
//   frag index = ((wave*8 + j)*16 + kb)*32 + lane     (16 bf16 = 32 B each)
// Wave `wave` owns hidden cols [32*wave, 32*wave+32); j = gate*2 + s picks
// gate and 16-col subtile.  K-pair held in vgpr v of lane group hi:
//   k0 = (v&3)*2 + hi*8 + (v>>2)*16 + kb*32
// ---------------------------------------------------------------------------
__global__ void lstm_prep_wh(const float* __restrict__ Wgh, const float* __restrict__ Wih,
                             const float* __restrict__ Wfh, const float* __restrict__ Woh,
                             bf16_t* __restrict__ whfrag) {
    int gid = blockIdx.x * blockDim.x + threadIdx.x;
    if (gid >= 16 * 8 * 16 * 32) return;
    int lane = gid & 31;
    int kb   = (gid >> 5) & 15;
    int j    = (gid >> 9) & 7;
    int wave = gid >> 12;
    int gate = j >> 1, s = j & 1;
    const float* W = (gate == 0) ? Wgh : (gate == 1) ? Wih : (gate == 2) ? Wfh : Woh;
    int n  = (wave * 2 + s) * 16 + (lane & 15);
    int hi = lane >> 4;
    union { bf16_t h[16]; uint4 q[2]; } f;
#pragma unroll
    for (int v = 0; v < 8; ++v) {
        int k0 = (v & 3) * 2 + hi * 8 + (v >> 2) * 16 + kb * 32;
        f.h[2 * v + 0] = (bf16_t)W[(k0 + 0) * LSTM_H + n];
        f.h[2 * v + 1] = (bf16_t)W[(k0 + 1) * LSTM_H + n];
    }
    uint4* dst = (uint4*)(whfrag + (size_t)gid * 16);
    dst[0] = f.q[0];
    dst[1] = f.q[1];
}

// ---------------------------------------------------------------------------
// Prep: input-projection weights Wx[g][E=10][N=512] + gate bias, padded to a
// single K=32 block.  Row k==10 carries the bias (A-side supplies 1.0 there).
//   frag index = (wave*8 + j)*32 + lane
// ---------------------------------------------------------------------------
__global__ void lstm_prep_wx(const float* __restrict__ Wgx, const float* __restrict__ bg,
                             const float* __restrict__ Wix, const float* __restrict__ bi,
                             const float* __restrict__ Wfx, const float* __restrict__ bfv,
                             const float* __restrict__ Wox, const float* __restrict__ bo,
                             bf16_t* __restrict__ wxfrag) {
    int gid = blockIdx.x * blockDim.x + threadIdx.x;
    if (gid >= 16 * 8 * 32) return;
    int lane = gid & 31;
    int j    = (gid >> 5) & 7;
    int wave = gid >> 8;
    int gate = j >> 1, s = j & 1;
    const float* W    = (gate == 0) ? Wgx : (gate == 1) ? Wix : (gate == 2) ? Wfx : Wox;
    const float* bias = (gate == 0) ? bg  : (gate == 1) ? bi  : (gate == 2) ? bfv : bo;
    int n  = (wave * 2 + s) * 16 + (lane & 15);
    int hi = lane >> 4;
    union { bf16_t h[16]; uint4 q[2]; } f;
#pragma unroll
    for (int v = 0; v < 8; ++v) {
        int k0 = (v & 3) * 2 + hi * 8 + (v >> 2) * 16;
#pragma unroll
        for (int p = 0; p < 2; ++p) {
            int k = k0 + p;
            float val = (k < LSTM_E) ? W[k * LSTM_H + n] : ((k == LSTM_E) ? bias[n] : 0.0f);
            f.h[2 * v + p] = (bf16_t)val;
        }
    }
    uint4* dst = (uint4*)(wxfrag + (size_t)gid * 16);
    dst[0] = f.q[0];
    dst[1] = f.q[1];
}

// ---------------------------------------------------------------------------
// Prep: padded embedding rows xepad[t][b][32] bf16 (k==10 is constant 1.0 so
// the x-projection WMMA also applies the gate bias).  Layout puts each
// step's 16 batch rows of a block contiguous (1 KB) for coalesced staging.
// ---------------------------------------------------------------------------
__global__ void lstm_prep_xe(const int* __restrict__ x, const float* __restrict__ emb,
                             bf16_t* __restrict__ xepad) {
    int gid = blockIdx.x * blockDim.x + threadIdx.x;   // = (t*512 + b)*32 + k
    if (gid >= LSTM_T * LSTM_B * 32) return;
    int k = gid & 31;
    int b = (gid >> 5) & (LSTM_B - 1);
    int t = gid >> 14;
    int tok = x[b * LSTM_T + t];
    float v = (k < LSTM_E) ? emb[tok * LSTM_E + k] : ((k == LSTM_E) ? 1.0f : 0.0f);
    xepad[gid] = (bf16_t)v;
}

__device__ __forceinline__ float sig_f(float x)  { return 1.0f / (1.0f + __expf(-x)); }
__device__ __forceinline__ float tanh_f(float x) { return 2.0f / (1.0f + __expf(-2.0f * x)) - 1.0f; }

// ---------------------------------------------------------------------------
// Main persistent LSTM kernel: 32 blocks x 16 rows of the batch, 512 threads
// (16 waves).  Wave w owns hidden cols [32w, 32w+32) for all 4 gates.
// c state lives in VGPRs; h (bf16) ping-pongs between two LDS buffers so one
// barrier per timestep suffices.  Next step's x rows are loaded at the top of
// the iteration and only stored to LDS after the WMMA loop, so the global
// load latency is hidden behind the matrix work.
// ---------------------------------------------------------------------------
__global__ __launch_bounds__(512) void lstm_main(
        const bf16_t* __restrict__ xepad,
        const bf16_t* __restrict__ whfrag, const bf16_t* __restrict__ wxfrag,
        const float* __restrict__ Wph, const float* __restrict__ bp,
        float* __restrict__ out) {
    __shared__ alignas(16) bf16_t h_lds[2][16 * LSTM_H];   // ping-pong, 16 x 512 each
    __shared__ alignas(16) bf16_t xe_lds[2][16 * 32];      // ping-pong, 16 x 32 each

    const int tid  = threadIdx.x;
    const int lane = tid & 31;
    const int wave = tid >> 5;
    const int row0 = blockIdx.x * 16;
    const int m    = lane & 15;
    const int hi   = lane >> 4;
    const bool pref = (tid < 64);

    // per-wave fragment bases: all (j, kb) offsets become load immediates
    const bf16_t* whb = whfrag + ((size_t)wave * 128 * 32 + lane) * 16;
    const bf16_t* wxb = wxfrag + ((size_t)wave * 8 * 32 + lane) * 16;

    for (int i = tid; i < 16 * LSTM_H; i += 512) h_lds[0][i] = (bf16_t)0.0f;
    if (pref) {  // stage x rows for t = 0
        const uint4* src = (const uint4*)xepad + (size_t)row0 * 4;
        ((uint4*)xe_lds[0])[tid] = src[tid];
    }

    float cst[2][8];
#pragma unroll
    for (int s = 0; s < 2; ++s)
#pragma unroll
        for (int r = 0; r < 8; ++r) cst[s][r] = 0.0f;

    __syncthreads();

    union Frag { v16bf v; uint4 q[2]; };

#pragma unroll 1
    for (int t = 0; t < LSTM_T; ++t) {
        const int pb = t & 1;
        const bf16_t* hrd = h_lds[pb];
        bf16_t*       hwr = h_lds[pb ^ 1];

        // --- issue next step's x-row load early (waves 0-1); stored later
        uint4 xnext;
        const bool doPref = pref && (t + 1 < LSTM_T);
        if (doPref) {
            const uint4* src = (const uint4*)xepad + ((size_t)(t + 1) * LSTM_B + row0) * 4;
            xnext = src[tid];
        }

        v8f acc[8] = {};
        Frag a, b;

        // --- x projection + bias: one WMMA per tile (reads xe_lds[pb];
        //     Wx fragments are loop-invariant and re-read from global each step)
        {
            const uint4* ap = (const uint4*)&xe_lds[pb][m * 32 + hi * 8];
            a.q[0] = ap[0];
            a.q[1] = ap[2];              // +32 bytes
#pragma unroll
            for (int j = 0; j < 8; ++j) {
                const uint4* bq = (const uint4*)(wxb + (size_t)j * 512);
                b.q[0] = bq[0];
                b.q[1] = bq[1];
                acc[j] = __builtin_amdgcn_wmma_f32_16x16x32_bf16(
                    false, a.v, false, b.v, (short)0, acc[j], false, false);
            }
        }

        // --- recurrent projection: K = 512 in 16 blocks of 32
#pragma unroll 4
        for (int kb = 0; kb < 16; ++kb) {
            const uint4* ap = (const uint4*)&hrd[m * LSTM_H + kb * 32 + hi * 8];
            a.q[0] = ap[0];
            a.q[1] = ap[2];              // +32 bytes
#pragma unroll
            for (int j = 0; j < 8; ++j) {
                const uint4* bq = (const uint4*)(whb + (size_t)(j * 16 + kb) * 512);
                b.q[0] = bq[0];
                b.q[1] = bq[1];
                acc[j] = __builtin_amdgcn_wmma_f32_16x16x32_bf16(
                    false, a.v, false, b.v, (short)0, acc[j], false, false);
            }
        }

        // --- deposit prefetched x rows (load latency hidden by WMMA loop)
        if (doPref) ((uint4*)xe_lds[pb ^ 1])[tid] = xnext;

        // --- gate nonlinearity, c update (registers), write new h (bf16)
#pragma unroll
        for (int s = 0; s < 2; ++s) {
#pragma unroll
            for (int r = 0; r < 8; ++r) {
                float pg = acc[0 + s][r];
                float pi = acc[2 + s][r];
                float pf = acc[4 + s][r];
                float po = acc[6 + s][r];
                float gg = tanh_f(pg);
                float ii = sig_f(pi);
                float ff = sig_f(pf);
                float oo = sig_f(po);
                float cn = gg * ii + cst[s][r] * ff;
                cst[s][r] = cn;
                float hn = tanh_f(cn) * oo;
                int mr = r + hi * 8;
                hwr[mr * LSTM_H + wave * 32 + s * 16 + m] = (bf16_t)hn;
            }
        }
        __syncthreads();   // publish h(t+1) and xe(t+1); guards ping-pong reuse
    }

    // --- final projection h_T @ Wph + bp  (16 rows x 10 classes per block)
    // T even -> final h is in buffer 0
    if (tid < 16 * LSTM_C) {
        int r = tid / LSTM_C, cc = tid % LSTM_C;
        float sum = bp[cc];
        for (int k = 0; k < LSTM_H; ++k)
            sum += (float)h_lds[LSTM_T & 1][r * LSTM_H + k] * Wph[k * LSTM_C + cc];
        out[(row0 + r) * LSTM_C + cc] = sum;
    }
}

extern "C" void kernel_launch(void* const* d_in, const int* in_sizes, int n_in,
                              void* d_out, int out_size, void* d_ws, size_t ws_size,
                              hipStream_t stream) {
    const int*   x   = (const int*)d_in[0];
    const float* emb = (const float*)d_in[1];
    const float* Wgx = (const float*)d_in[2];
    const float* Wgh = (const float*)d_in[3];
    const float* bg  = (const float*)d_in[4];
    const float* Wix = (const float*)d_in[5];
    const float* Wih = (const float*)d_in[6];
    const float* bi  = (const float*)d_in[7];
    const float* Wfx = (const float*)d_in[8];
    const float* Wfh = (const float*)d_in[9];
    const float* bfv = (const float*)d_in[10];
    const float* Wox = (const float*)d_in[11];
    const float* Woh = (const float*)d_in[12];
    const float* bo  = (const float*)d_in[13];
    const float* Wph = (const float*)d_in[14];
    const float* bp  = (const float*)d_in[15];

    bf16_t* whfrag = (bf16_t*)d_ws;                          // 2 MB
    bf16_t* wxfrag = whfrag + (size_t)4 * LSTM_H * LSTM_H;   // 128 KB
    bf16_t* xepad  = wxfrag + (size_t)16 * 8 * 32 * 16;      // 8 MB

    lstm_prep_wh<<<256, 256, 0, stream>>>(Wgh, Wih, Wfh, Woh, whfrag);
    lstm_prep_wx<<<16, 256, 0, stream>>>(Wgx, bg, Wix, bi, Wfx, bfv, Wox, bo, wxfrag);
    lstm_prep_xe<<<(LSTM_T * LSTM_B * 32) / 256, 256, 0, stream>>>(x, emb, xepad);
    lstm_main<<<32, 512, 0, stream>>>(xepad, whfrag, wxfrag, Wph, bp, (float*)d_out);
}
